// WorldModel_43946105373238
// MI455X (gfx1250) — compile-verified, matrix operand order checked by reading, not run on hardware
//
#include <hip/hip_runtime.h>
#include <hip/hip_bf16.h>

// ---------------------------------------------------------------------------
// DreamerV2-style world model forward for MI455X (gfx1250).
// bf16 WMMA (v_wmma_f32_16x16x32_bf16) GEMMs, fp32 accumulation.
// Double-buffered LDS staging; async global->LDS copies when the toolchain
// exposes the gfx1250 builtins (guarded), else a reg-pipelined sync path.
// ---------------------------------------------------------------------------

#define B_   256
#define T_   256
#define OBS_ 256
#define ACT_ 32
#define HID_ 1024
#define DET_ 1024
#define STO_ 128
#define G3_  3072
#define BT_  (B_*T_)
#define KAS_ 192   // as_W K (=160) zero-padded to a multiple of 64

typedef unsigned short u16;
typedef __attribute__((ext_vector_type(8)))  u16    u16x8;
typedef __attribute__((ext_vector_type(8)))  __bf16 bf16x8;
typedef __attribute__((ext_vector_type(16))) __bf16 bf16x16;
typedef __attribute__((ext_vector_type(8)))  float  f32x8;
typedef int v4i_g __attribute__((vector_size(16)));   // GCC-style, matches builtin

#define AS1 __attribute__((address_space(1)))
#define AS3 __attribute__((address_space(3)))

#if defined(__has_builtin)
#if __has_builtin(__builtin_amdgcn_global_load_async_to_lds_b128) && \
    __has_builtin(__builtin_amdgcn_s_wait_asynccnt)
#define USE_ASYNC_LDS 1
#endif
#endif
#ifndef USE_ASYNC_LDS
#define USE_ASYNC_LDS 0
#endif

__device__ __forceinline__ u16 f2b(float f) {
  unsigned u = __float_as_uint(f);
  u += 0x7FFFu + ((u >> 16) & 1u);      // round-to-nearest-even
  return (u16)(u >> 16);
}
__device__ __forceinline__ float b2f(u16 h) {
  return __uint_as_float(((unsigned)h) << 16);
}
__device__ __forceinline__ float elu_f(float x)     { return x > 0.f ? x : (__expf(x) - 1.f); }
__device__ __forceinline__ float sigmoid_f(float x) { return 1.f / (1.f + __expf(-x)); }
__device__ __forceinline__ float softplus_f(float x){ return fmaxf(x, 0.f) + log1pf(__expf(-fabsf(x))); }

// ---------------------------------------------------------------------------
// Generic tiled bf16 WMMA GEMM:  C[M,N] = act(A[M,K] * W[N,K]^T + bias[N])
//  - 256 threads (8 wave32), 128x64 output tile, BK=64 (2 wmma K slabs)
//  - double-buffered LDS; async global->LDS if available
//  - optional fused per-column squared-error reduction vs Tgt (obs loss)
// Requires: M%128==0, N%64==0, K%64==0 (true for every call site here).
// ---------------------------------------------------------------------------
#define BM  128
#define BN  64
#define BKK 64
#define LDT 72   // padded LDS row stride (ushorts): 144B rows, 16B aligned

template<int ACTV, bool STF, bool STB, bool LOSS>
__global__ __launch_bounds__(256) void gemm_bf16_k(
    const u16*  __restrict__ A, const u16* __restrict__ W,
    const float* __restrict__ bias,
    float* __restrict__ Cf, u16* __restrict__ Cb,
    const float* __restrict__ Tgt, float* __restrict__ colAcc,
    int M, int N, int K)
{
  __shared__ u16   As[2][BM * LDT];
  __shared__ u16   Ws[2][BN * LDT];
  __shared__ float colsum[BN];

  const int tid  = threadIdx.x;
  const int wave = tid >> 5;
  const int lane = tid & 31;
  const int lo   = lane & 15;
  const int hi   = lane >> 4;
  const int m0   = blockIdx.y * BM;
  const int n0   = blockIdx.x * BN;

  if (LOSS && tid < BN) colsum[tid] = 0.f;

  f32x8 acc[4] = {};
  const int nSlab = K / BKK;

  // One WMMA macro-step over the current LDS buffer (8 wmma / wave).
  auto compute = [&](int cur) {
    const u16* Ab = As[cur];
    const u16* Wb = Ws[cur];
    const int aRow = (wave * 16 + lo) * LDT;
    #pragma unroll
    for (int kk = 0; kk < BKK; kk += 32) {
      bf16x8 al = *(const bf16x8*)(&Ab[aRow + kk + hi * 8]);
      bf16x8 ah = *(const bf16x8*)(&Ab[aRow + kk + 16 + hi * 8]);
      bf16x16 af = __builtin_shufflevector(al, ah,
          0,1,2,3,4,5,6,7,8,9,10,11,12,13,14,15);
      #pragma unroll
      for (int s = 0; s < 4; ++s) {
        const int wRow = (s * 16 + lo) * LDT;
        bf16x8 bl = *(const bf16x8*)(&Wb[wRow + kk + hi * 8]);
        bf16x8 bh = *(const bf16x8*)(&Wb[wRow + kk + 16 + hi * 8]);
        bf16x16 bq = __builtin_shufflevector(bl, bh,
            0,1,2,3,4,5,6,7,8,9,10,11,12,13,14,15);
        acc[s] = __builtin_amdgcn_wmma_f32_16x16x32_bf16(
            false, af, false, bq, (short)0, acc[s], false, false);
      }
    }
  };

#if USE_ASYNC_LDS
  // -------- async path: staging rides ASYNCcnt, decoupled from DScnt ------
  auto stage_async = [&](int k0, int nb) {
    u16* as = As[nb];
    u16* ws = Ws[nb];
    #pragma unroll
    for (int c = 0; c < 4; ++c) {          // A: 128 rows x 64 halves
      int idx = tid + c * 256;
      int r = idx >> 3, cc = (idx & 7) * 8;
      __builtin_amdgcn_global_load_async_to_lds_b128(
          (AS1 v4i_g*)&A[(size_t)(m0 + r) * K + k0 + cc],
          (AS3 v4i_g*)&as[r * LDT + cc], 0, 0);
    }
    #pragma unroll
    for (int c = 0; c < 2; ++c) {          // W: 64 rows x 64 halves
      int idx = tid + c * 256;
      int r = idx >> 3, cc = (idx & 7) * 8;
      __builtin_amdgcn_global_load_async_to_lds_b128(
          (AS1 v4i_g*)&W[(size_t)(n0 + r) * K + k0 + cc],
          (AS3 v4i_g*)&ws[r * LDT + cc], 0, 0);
    }
  };
  stage_async(0, 0);
  for (int i = 0; i < nSlab; ++i) {
    const int cur = i & 1;
    __builtin_amdgcn_s_wait_asynccnt(0);   // my slab-i copies landed in LDS
    __syncthreads();                       // everyone's landed; prev compute done
    if (i + 1 < nSlab) stage_async((i + 1) * BKK, cur ^ 1);
    if (i + 2 < nSlab)                     // pull slab i+2 weights toward L2
      __builtin_prefetch(&W[(size_t)(n0 + (tid >> 3)) * K + (i + 2) * BKK], 0, 1);
    compute(cur);
  }
#else
  // -------- sync fallback: global->VGPR overlaps WMMA, ds_store after -----
  {
    #pragma unroll
    for (int c = 0; c < 4; ++c) {
      int idx = tid + c * 256;
      int r = idx >> 3, cc = (idx & 7) * 8;
      *(u16x8*)(&As[0][r * LDT + cc]) = *(const u16x8*)(&A[(size_t)(m0 + r) * K + cc]);
    }
    #pragma unroll
    for (int c = 0; c < 2; ++c) {
      int idx = tid + c * 256;
      int r = idx >> 3, cc = (idx & 7) * 8;
      *(u16x8*)(&Ws[0][r * LDT + cc]) = *(const u16x8*)(&W[(size_t)(n0 + r) * K + cc]);
    }
  }
  for (int i = 0; i < nSlab; ++i) {
    const int cur = i & 1;
    __syncthreads();
    const bool more = (i + 1 < nSlab);
    const int k1 = (i + 1) * BKK;
    u16x8 ra[4], rw[2];
    if (more) {
      #pragma unroll
      for (int c = 0; c < 4; ++c) {
        int idx = tid + c * 256;
        int r = idx >> 3, cc = (idx & 7) * 8;
        ra[c] = *(const u16x8*)(&A[(size_t)(m0 + r) * K + k1 + cc]);
      }
      #pragma unroll
      for (int c = 0; c < 2; ++c) {
        int idx = tid + c * 256;
        int r = idx >> 3, cc = (idx & 7) * 8;
        rw[c] = *(const u16x8*)(&W[(size_t)(n0 + r) * K + k1 + cc]);
      }
    }
    compute(cur);
    if (more) {
      u16* as = As[cur ^ 1];
      u16* ws = Ws[cur ^ 1];
      #pragma unroll
      for (int c = 0; c < 4; ++c) {
        int idx = tid + c * 256;
        int r = idx >> 3, cc = (idx & 7) * 8;
        *(u16x8*)(&as[r * LDT + cc]) = ra[c];
      }
      #pragma unroll
      for (int c = 0; c < 2; ++c) {
        int idx = tid + c * 256;
        int r = idx >> 3, cc = (idx & 7) * 8;
        *(u16x8*)(&ws[r * LDT + cc]) = rw[c];
      }
    }
  }
#endif

  // Epilogue: C/D layout -> lane holds col (n0+s*16+lo), rows hi*8+v
  #pragma unroll
  for (int s = 0; s < 4; ++s) {
    const int col  = n0 + s * 16 + lo;
    const float bv = bias ? bias[col] : 0.f;
    float se = 0.f;
    #pragma unroll
    for (int v = 0; v < 8; ++v) {
      const int row = m0 + wave * 16 + hi * 8 + v;
      float val = acc[s][v] + bv;
      if (ACTV == 1) val = elu_f(val);
      const size_t o = (size_t)row * N + col;
      if (STF) Cf[o] = val;
      if (STB) Cb[o] = f2b(val);
      if (LOSS) { float d = val - Tgt[o]; se += d * d; }
    }
    if (LOSS) atomicAdd(&colsum[s * 16 + lo], se);
  }
  if (LOSS) {
    __syncthreads();
    if (tid < BN) atomicAdd(&colAcc[n0 + tid], colsum[tid]);
  }
}

// ---------------------------------------------------------------------------
// Elementwise / helper kernels
// ---------------------------------------------------------------------------
__global__ void cvt_bf16_k(const float* __restrict__ s, u16* __restrict__ d, size_t n) {
  size_t i = (size_t)blockIdx.x * 256 + threadIdx.x;
  if (i < n) d[i] = f2b(s[i]);
}
// Convert (N,Ksrc) fp32 -> (N,Kpad) bf16, zero-padding K tail.
__global__ void cvt_pad_k(const float* __restrict__ s, u16* __restrict__ d,
                          int N, int Ksrc, int Kpad) {
  int i = blockIdx.x * 256 + threadIdx.x;
  if (i >= N * Kpad) return;
  int n = i / Kpad, k = i % Kpad;
  d[i] = (k < Ksrc) ? f2b(s[(size_t)n * Ksrc + k]) : (u16)0;
}
__global__ void zero_f32_k(float* p, int n) {
  int i = blockIdx.x * 256 + threadIdx.x; if (i < n) p[i] = 0.f;
}
__global__ void zero_u16_k(u16* p, int n) {
  int i = blockIdx.x * 256 + threadIdx.x; if (i < n) p[i] = 0;
}

// as_in = [a_t | stoch | zero-pad]  (B x KAS_, bf16)
__global__ void pack_as_k(const float* __restrict__ actions, const u16* __restrict__ stoch,
                          u16* __restrict__ as_in, int t) {
  int i = blockIdx.x * 256 + threadIdx.x;
  if (i >= B_ * KAS_) return;
  int b = i / KAS_, c = i % KAS_;
  u16 v;
  if (c < ACT_)             v = f2b(actions[((size_t)b * T_ + t) * ACT_ + c]);
  else if (c < ACT_ + STO_) v = stoch[b * STO_ + (c - ACT_)];
  else                      v = 0;
  as_in[i] = v;
}

// GRU gate math; also stages deter into dec_in and pos_in = [deter | e_t]
__global__ void gate_k(const float* __restrict__ gi, const float* __restrict__ gh,
                       float* __restrict__ deterf, u16* __restrict__ deterb,
                       u16* __restrict__ dec_in, u16* __restrict__ pos_in,
                       const u16* __restrict__ ebuf, int t) {
  int i = blockIdx.x * 256 + threadIdx.x;
  if (i >= B_ * DET_) return;
  int b = i >> 10, j = i & (DET_ - 1);
  size_t g = (size_t)b * G3_;
  float r  = sigmoid_f(gi[g + j]            + gh[g + j]);
  float z  = sigmoid_f(gi[g + DET_ + j]     + gh[g + DET_ + j]);
  float nn = tanhf   (gi[g + 2 * DET_ + j] + r * gh[g + 2 * DET_ + j]);
  float d  = deterf[i];
  d = (1.f - z) * nn + z * d;
  deterf[i] = d;
  u16 db = f2b(d);
  deterb[i] = db;
  dec_in[((size_t)b * T_ + t) * (STO_ + DET_) + STO_ + j] = db;
  pos_in[(size_t)b * (DET_ + HID_) + j]        = db;
  pos_in[(size_t)b * (DET_ + HID_) + DET_ + j] = ebuf[((size_t)b * T_ + t) * HID_ + j];
}

__global__ void prior_k(const float* __restrict__ pl, float* __restrict__ pmean,
                        float* __restrict__ pstd, int t) {
  int i = blockIdx.x * 256 + threadIdx.x;
  if (i >= B_ * STO_) return;
  int b = i >> 7, s = i & (STO_ - 1);
  size_t o = ((size_t)b * T_ + t) * STO_ + s;
  pmean[o] = pl[b * 2 * STO_ + s];
  pstd [o] = softplus_f(pl[b * 2 * STO_ + STO_ + s]);
}

__global__ void post_k(const float* __restrict__ ql, const float* __restrict__ noise,
                       u16* __restrict__ stochb, u16* __restrict__ dec_in,
                       float* __restrict__ qmean, float* __restrict__ qstd, int t) {
  int i = blockIdx.x * 256 + threadIdx.x;
  if (i >= B_ * STO_) return;
  int b = i >> 7, s = i & (STO_ - 1);
  float qm = ql[b * 2 * STO_ + s];
  float qs = softplus_f(ql[b * 2 * STO_ + STO_ + s]);
  float st = qm + qs * noise[((size_t)b * T_ + t) * STO_ + s];
  u16 sb = f2b(st);
  stochb[i] = sb;
  dec_in[((size_t)b * T_ + t) * (STO_ + DET_) + s] = sb;
  if (t == T_ - 1) { qmean[i] = qm; qstd[i] = qs; }   // last-step posterior
}

// KL: reference broadcasts post (B,STO) against prior (B,T,STO) -> post[t,s]
__global__ void kl_k(const float* __restrict__ pm, const float* __restrict__ ps,
                     const float* __restrict__ qm, const float* __restrict__ qs,
                     float* __restrict__ acc) {
  int i = blockIdx.x * 256 + threadIdx.x;
  if (i >= B_ * T_) return;
  int b = i >> 8, t = i & (T_ - 1);
  size_t base = ((size_t)b * T_ + t) * STO_;
  float sum = 0.f;
  for (int s = 0; s < STO_; ++s) {
    float pstd_v = ps[base + s];
    float pm_v   = pm[base + s];
    float qstd_v = qs[t * STO_ + s];
    float qm_v   = qm[t * STO_ + s];
    float dm = qm_v - pm_v;
    sum += __logf(pstd_v / qstd_v)
         + (qstd_v * qstd_v + dm * dm) / (2.f * pstd_v * pstd_v) - 0.5f;
  }
  atomicAdd(acc, fmaxf(sum, 1.f));
}

// Reward head (N=1): one wave32 per row, shuffle reduction, fused MSE
__global__ __launch_bounds__(256) void rdot_k(
    const u16* __restrict__ h, const float* __restrict__ w3,
    const float* __restrict__ b3, const float* __restrict__ rewards,
    float* __restrict__ acc) {
  int wave = threadIdx.x >> 5, lane = threadIdx.x & 31;
  int row = blockIdx.x * 8 + wave;
  if (row >= BT_) return;
  float s = 0.f;
  for (int k = lane; k < HID_; k += 32)
    s += b2f(h[(size_t)row * HID_ + k]) * w3[k];
  #pragma unroll
  for (int o = 16; o > 0; o >>= 1) s += __shfl_xor(s, o, 32);
  if (lane == 0) {
    float d = (s + b3[0]) - rewards[row];
    atomicAdd(acc, d * d);
  }
}

__global__ void finalize_k(const float* __restrict__ accObs, const float* __restrict__ accR,
                           const float* __restrict__ accKl, float* __restrict__ out,
                           int out_size) {
  int c = threadIdx.x;
  float rr = accR[0] * (1.f / (float)BT_);
  float kl = accKl[0] * (1.f / (float)BT_);
  if (c < OBS_) {
    float ol = accObs[c] * (1.f / (float)BT_);
    if (c < out_size)        out[c] = ol + rr + kl;        // total (shape OBS)
    if (OBS_ + c < out_size) out[OBS_ + c] = ol;           // recon_obs_loss
  }
  if (c == 0) {
    if (2 * OBS_ < out_size)     out[2 * OBS_] = rr;       // recon_reward_loss
    if (2 * OBS_ + 1 < out_size) out[2 * OBS_ + 1] = kl;   // kl_loss
  }
}

// ---------------------------------------------------------------------------
// Host orchestration
// ---------------------------------------------------------------------------
extern "C" void kernel_launch(void* const* d_in, const int* in_sizes, int n_in,
                              void* d_out, int out_size, void* d_ws, size_t ws_size,
                              hipStream_t stream)
{
  (void)in_sizes; (void)n_in; (void)ws_size;
  const float* obs     = (const float*)d_in[0];
  const float* actions = (const float*)d_in[1];
  const float* rewards = (const float*)d_in[2];
  const float* noise   = (const float*)d_in[3];
  const float* P[32];
  for (int i = 0; i < 32; ++i) P[i] = (const float*)d_in[4 + i];

  enum {
    enc_W1, enc_b1, enc_W2, enc_b2, enc_W3, enc_b3,
    as_W, as_b, gru_Wih, gru_bih, gru_Whh, gru_bhh,
    pri_W1, pri_b1, pri_W2, pri_b2, pos_W1, pos_b1, pos_W2, pos_b2,
    od_W1, od_b1, od_W2, od_b2, od_W3, od_b3,
    rd_W1, rd_b1, rd_W2, rd_b2, rd_W3, rd_b3
  };

  char* base = (char*)d_ws;
  size_t off = 0;
  #define AU16(name, n) u16*   name = (u16*)  (base + off); off += (((size_t)(n)) * 2 + 255) & ~(size_t)255;
  #define AF32(name, n) float* name = (float*)(base + off); off += (((size_t)(n)) * 4 + 255) & ~(size_t)255;

  // bf16 weights
  AU16(wEnc1, HID_ * OBS_)          AU16(wEnc2, HID_ * HID_)
  AU16(wEnc3, HID_ * HID_)          AU16(wAs,   HID_ * KAS_)
  AU16(wIh,   G3_ * HID_)           AU16(wHh,   G3_ * DET_)
  AU16(wPri1, HID_ * DET_)          AU16(wPri2, 2 * STO_ * HID_)
  AU16(wPos1, HID_ * (DET_ + HID_)) AU16(wPos2, 2 * STO_ * HID_)
  AU16(wOd1,  HID_ * (STO_ + DET_)) AU16(wOd2, HID_ * HID_)
  AU16(wOd3,  OBS_ * HID_)
  AU16(wRd1,  HID_ * (STO_ + DET_)) AU16(wRd2, HID_ * HID_)
  // bf16 activations
  AU16(obs_bf,  (size_t)BT_ * OBS_)
  AU16(bufA,    (size_t)BT_ * HID_)
  AU16(bufB,    (size_t)BT_ * HID_)
  AU16(bufE,    (size_t)BT_ * HID_)
  AU16(dec_in,  (size_t)BT_ * (STO_ + DET_))
  AU16(pos_in,  (size_t)B_ * (DET_ + HID_))
  AU16(as_in,   B_ * KAS_)
  AU16(x_bf,    B_ * HID_)
  AU16(deter_b, B_ * DET_)
  AU16(stoch_b, B_ * STO_)
  AU16(hpri,    B_ * HID_)
  AU16(hpos,    B_ * HID_)
  // fp32 state / stats
  AF32(gi, B_ * G3_)          AF32(gh, B_ * G3_)
  AF32(deter_f, B_ * DET_)
  AF32(pl, B_ * 2 * STO_)     AF32(ql, B_ * 2 * STO_)
  AF32(prior_mean, (size_t)BT_ * STO_)
  AF32(prior_std,  (size_t)BT_ * STO_)
  AF32(post_mean, B_ * STO_)  AF32(post_std, B_ * STO_)
  AF32(acc_obs, OBS_)         AF32(accRK, 2)

  #define CVT(src, dst, n) \
    cvt_bf16_k<<<dim3((unsigned)(((size_t)(n) + 255) / 256)), 256, 0, stream>>>(src, dst, (size_t)(n));
  CVT(P[enc_W1], wEnc1, HID_ * OBS_)           CVT(P[enc_W2], wEnc2, HID_ * HID_)
  CVT(P[enc_W3], wEnc3, HID_ * HID_)
  cvt_pad_k<<<(HID_ * KAS_ + 255) / 256, 256, 0, stream>>>(P[as_W], wAs, HID_, ACT_ + STO_, KAS_);
  CVT(P[gru_Wih], wIh, G3_ * HID_)             CVT(P[gru_Whh], wHh, G3_ * DET_)
  CVT(P[pri_W1], wPri1, HID_ * DET_)           CVT(P[pri_W2], wPri2, 2 * STO_ * HID_)
  CVT(P[pos_W1], wPos1, HID_ * (DET_ + HID_))  CVT(P[pos_W2], wPos2, 2 * STO_ * HID_)
  CVT(P[od_W1], wOd1, HID_ * (STO_ + DET_))    CVT(P[od_W2], wOd2, HID_ * HID_)
  CVT(P[od_W3], wOd3, OBS_ * HID_)
  CVT(P[rd_W1], wRd1, HID_ * (STO_ + DET_))    CVT(P[rd_W2], wRd2, HID_ * HID_)
  CVT(obs, obs_bf, (size_t)BT_ * OBS_)

  zero_f32_k<<<1, 256, 0, stream>>>(acc_obs, OBS_);
  zero_f32_k<<<1, 256, 0, stream>>>(accRK, 2);
  zero_f32_k<<<(B_ * DET_ + 255) / 256, 256, 0, stream>>>(deter_f, B_ * DET_);
  zero_u16_k<<<(B_ * DET_ + 255) / 256, 256, 0, stream>>>(deter_b, B_ * DET_);
  zero_u16_k<<<(B_ * STO_ + 255) / 256, 256, 0, stream>>>(stoch_b, B_ * STO_);

  #define GEMM(ACTV, STF, STB, LOSSV, Ap, Wp, Bp, CfP, CbP, TgtP, AccP, M, N, K)           \
    gemm_bf16_k<ACTV, STF, STB, LOSSV>                                                     \
        <<<dim3((unsigned)((N) / BN), (unsigned)((M) / BM)), 256, 0, stream>>>(            \
            Ap, Wp, Bp, CfP, CbP, TgtP, AccP, M, N, K);

  // ---- Encoder: e = enc3(elu(enc2(elu(enc1(obs))))) --------------------
  GEMM(1, false, true, false, obs_bf, wEnc1, P[enc_b1], nullptr, bufA, nullptr, nullptr, BT_, HID_, OBS_)
  GEMM(1, false, true, false, bufA,   wEnc2, P[enc_b2], nullptr, bufB, nullptr, nullptr, BT_, HID_, HID_)
  GEMM(0, false, true, false, bufB,   wEnc3, P[enc_b3], nullptr, bufE, nullptr, nullptr, BT_, HID_, HID_)

  // ---- Recurrent scan ---------------------------------------------------
  for (int t = 0; t < T_; ++t) {
    pack_as_k<<<(B_ * KAS_ + 255) / 256, 256, 0, stream>>>(actions, stoch_b, as_in, t);
    GEMM(0, false, true, false, as_in,   wAs,  P[as_b],     nullptr, x_bf, nullptr, nullptr, B_, HID_, KAS_)
    GEMM(0, true, false, false, x_bf,    wIh,  P[gru_bih],  gi, nullptr, nullptr, nullptr, B_, G3_, HID_)
    GEMM(0, true, false, false, deter_b, wHh,  P[gru_bhh],  gh, nullptr, nullptr, nullptr, B_, G3_, DET_)
    gate_k<<<(B_ * DET_ + 255) / 256, 256, 0, stream>>>(gi, gh, deter_f, deter_b, dec_in, pos_in, bufE, t);
    GEMM(1, false, true, false, deter_b, wPri1, P[pri_b1], nullptr, hpri, nullptr, nullptr, B_, HID_, DET_)
    GEMM(0, true, false, false, hpri,    wPri2, P[pri_b2], pl, nullptr, nullptr, nullptr, B_, 2 * STO_, HID_)
    prior_k<<<(B_ * STO_ + 255) / 256, 256, 0, stream>>>(pl, prior_mean, prior_std, t);
    GEMM(1, false, true, false, pos_in,  wPos1, P[pos_b1], nullptr, hpos, nullptr, nullptr, B_, HID_, DET_ + HID_)
    GEMM(0, true, false, false, hpos,    wPos2, P[pos_b2], ql, nullptr, nullptr, nullptr, B_, 2 * STO_, HID_)
    post_k<<<(B_ * STO_ + 255) / 256, 256, 0, stream>>>(ql, noise, stoch_b, dec_in, post_mean, post_std, t);
  }

  // ---- Decoders ---------------------------------------------------------
  GEMM(1, false, true, false, dec_in, wOd1, P[od_b1], nullptr, bufA, nullptr, nullptr, BT_, HID_, STO_ + DET_)
  GEMM(1, false, true, false, bufA,   wOd2, P[od_b2], nullptr, bufB, nullptr, nullptr, BT_, HID_, HID_)
  GEMM(0, false, false, true, bufB,   wOd3, P[od_b3], nullptr, nullptr, obs, acc_obs, BT_, OBS_, HID_)

  GEMM(1, false, true, false, dec_in, wRd1, P[rd_b1], nullptr, bufA, nullptr, nullptr, BT_, HID_, STO_ + DET_)
  GEMM(1, false, true, false, bufA,   wRd2, P[rd_b2], nullptr, bufB, nullptr, nullptr, BT_, HID_, HID_)
  rdot_k<<<BT_ / 8, 256, 0, stream>>>(bufB, P[rd_W3], P[rd_b3], rewards, &accRK[0]);

  kl_k<<<(BT_ + 255) / 256, 256, 0, stream>>>(prior_mean, prior_std, post_mean, post_std, &accRK[1]);
  finalize_k<<<1, 256, 0, stream>>>(acc_obs, &accRK[0], &accRK[1], (float*)d_out, out_size);

  #undef AU16
  #undef AF32
  #undef CVT
  #undef GEMM
}
